// MultiHeadAttention_87024627352342
// MI455X (gfx1250) — matrix operand order repaired and checked by reading, not compiled, not run on hardware
//
#include <hip/hip_runtime.h>
#include <hip/hip_bf16.h>

// MultiHeadAttention on MI455X (gfx1250).
// B=4, Q=K=2048, D=512, H=8, DH=64.
// - all matmuls: v_wmma_f32_16x16x32_bf16 (bf16 in, f32 accumulate)
// - attention K/V tiles staged LDS via global_load_async_to_lds_b128,
//   double-buffered, synchronized with s_wait_asynccnt (ASYNCcnt).

typedef __bf16 v16bf __attribute__((ext_vector_type(16)));
typedef float  v8f   __attribute__((ext_vector_type(8)));

#define BATCH 4
#define SEQ   2048
#define DMODEL 512
#define NHEAD 8
#define DHEAD 64
#define MASK_VAL (-1.0e6f)

static __device__ inline v8f zero8() {
  v8f z = {0.f, 0.f, 0.f, 0.f, 0.f, 0.f, 0.f, 0.f};
  return z;
}

static __device__ inline v8f wmma_bf16(v16bf a, v16bf b, v8f c) {
  // 8 args: (neg_a, A, neg_b, B, c_mod, C, reuse_a, reuse_b)
  return __builtin_amdgcn_wmma_f32_16x16x32_bf16(false, a, false, b, (short)0, c,
                                                 false, false);
}

// A-matrix fragment, 16x32 bf16 (contiguous k order used by this code):
//   lane l (m=l&15, half=l>>4) holds row M=m,
//   element e -> K = kbase + 16*half + e
static __device__ inline v16bf load_a_frag(const __bf16* rowptr, int kbase, int half) {
  v16bf a;
  const __bf16* p0 = rowptr + kbase + 16 * half;
#pragma unroll
  for (int e = 0; e < 16; ++e) {
    a[e] = p0[e];
  }
  return a;
}

// B-matrix fragment, 32x16 bf16: lane l holds column N=(l&15),
// element e -> contraction K = 16*half + e (contiguous in memory at p).
static __device__ inline v16bf load_b_frag(const __bf16* p) {
  v16bf b;
#pragma unroll
  for (int e = 0; e < 16; ++e) b[e] = p[e];
  return b;
}

// ---- CDNA5 async LDS staging (ASYNCcnt path, ISA §15.18 / 08_async_tensor) ----
static __device__ inline void async_load_b128(unsigned lds_off, const void* gaddr) {
  // vdst = LDS byte address, vaddr = 64-bit global address, saddr = off
  asm volatile("global_load_async_to_lds_b128 %0, %1, off"
               :
               : "v"(lds_off), "v"(gaddr)
               : "memory");
}
static __device__ inline unsigned lds_off_of(const void* p) {
  return (unsigned)(uintptr_t)p;  // low 32 bits of flat addr == LDS offset
}

__global__ void __launch_bounds__(256) cvt_f32_bf16_kernel(const float* __restrict__ in,
                                                           __bf16* __restrict__ out, int n) {
  int i = blockIdx.x * blockDim.x + threadIdx.x;
  if (i < n) out[i] = (__bf16)in[i];
}

// Y = X @ W.T, X:[Nrows,512] bf16 row-major, W:[512,512] bf16 row-major.
// One wave computes a 16x64 output tile (A fragment reused across 4 WMMAs).
// mode 0: write bf16 split-head [B, H, S, DH]   (projection outputs)
// mode 1: write f32 flat [Nrows, 512]           (final output)
__global__ void __launch_bounds__(32) gemm_xwT_kernel(const __bf16* __restrict__ X,
                                                      const __bf16* __restrict__ W,
                                                      __bf16* __restrict__ outb,
                                                      float* __restrict__ outf, int mode) {
  const int lane = threadIdx.x;
  const int m = lane & 15;
  const int half = lane >> 4;
  const int o0 = blockIdx.x * 64;  // output column super-tile (4 x 16)
  const int n0 = blockIdx.y * 16;  // output row tile

  const __bf16* xrow = X + (size_t)(n0 + m) * DMODEL;

  v8f c[4];
#pragma unroll
  for (int ob = 0; ob < 4; ++ob) c[ob] = zero8();

#pragma unroll 2
  for (int kk = 0; kk < DMODEL / 32; ++kk) {
    v16bf a = load_a_frag(xrow, kk * 32, half);
#pragma unroll
    for (int ob = 0; ob < 4; ++ob) {
      const __bf16* wrow = W + (size_t)(o0 + ob * 16 + m) * DMODEL;
      v16bf bf = load_b_frag(wrow + kk * 32 + 16 * half);
      c[ob] = wmma_bf16(a, bf, c[ob]);
    }
  }

#pragma unroll
  for (int ob = 0; ob < 4; ++ob) {
#pragma unroll
    for (int r = 0; r < 8; ++r) {
      const int row = n0 + r + 8 * half;  // C layout: VGPR r, lanes16-31 -> M=r+8
      const int col = o0 + ob * 16 + m;   // N = lane & 15
      if (mode == 0) {
        const int bidx = row >> 11;       // / SEQ
        const int s = row & (SEQ - 1);
        const int hidx = col >> 6;        // / DHEAD
        const int d = col & (DHEAD - 1);
        outb[(((size_t)bidx * NHEAD + hidx) * SEQ + s) * DHEAD + d] = (__bf16)c[ob][r];
      } else {
        outf[(size_t)row * DMODEL + col] = c[ob][r];
      }
    }
  }
}

// Flash attention: one wave per (b, h, 32-row q-tile). 32 keys per iteration.
// K/V tiles (4 KB each) staged in LDS by the async engine, double-buffered.
// Qp/Kp/Vp are bf16 [B, H, S, DH]. Output written bf16 [B, Q, H*DH].
#define KTILE 32
#define NTILES (SEQ / KTILE)  // 64
#define TILE_ELTS (KTILE * DHEAD)  // 2048 bf16 = 4 KB

__global__ void __launch_bounds__(32) flash_attn_kernel(const __bf16* __restrict__ Qp,
                                                        const __bf16* __restrict__ Kp,
                                                        const __bf16* __restrict__ Vp,
                                                        const int* __restrict__ valid_lens,
                                                        __bf16* __restrict__ attn_out) {
  const int lane = threadIdx.x;
  const int m = lane & 15;
  const int half = lane >> 4;
  const int bh = blockIdx.y;        // B*H = 32
  const int b = bh >> 3;
  const int h = bh & (NHEAD - 1);
  const int qt = blockIdx.x;        // Q/32 = 64
  const int vlen = valid_lens[b];
  const float scale = 0.125f;       // 1/sqrt(DH)

  const __bf16* qbase = Qp + ((size_t)bh * SEQ + qt * 32) * DHEAD;
  const char* kbytes = (const char*)(Kp + (size_t)bh * SEQ * DHEAD);
  const char* vbytes = (const char*)(Vp + (size_t)bh * SEQ * DHEAD);

  __shared__ __bf16 sK[2][TILE_ELTS];  // double-buffered K tile (row-major [32][64])
  __shared__ __bf16 sV[2][TILE_ELTS];  // double-buffered V tile (row-major [32][64])
  __shared__ float  sS[32 * 32];       // score tile (f32)
  __shared__ __bf16 sP[32 * 32];       // softmax probabilities
  __shared__ float  sRow[32];          // per-row rescale / inverse-sum broadcast

  // Issue one tile's worth of async B128 transfers: 4 KB K + 4 KB V = 16 ops/wave.
  auto issue_tile = [&](int kt, int buf) {
    const size_t tile_bytes = (size_t)kt * KTILE * DHEAD * 2;
#pragma unroll
    for (int i = 0; i < 8; ++i) {
      const unsigned off = i * 512 + lane * 16;  // bytes within 4 KB tile
      async_load_b128(lds_off_of((const char*)&sK[buf][0] + off),
                      kbytes + tile_bytes + off);
      async_load_b128(lds_off_of((const char*)&sV[buf][0] + off),
                      vbytes + tile_bytes + off);
    }
  };

  // Q A-fragments for 32 rows (2 M-tiles x 2 contraction-32 chunks), in VGPRs.
  v16bf aQ[2][2];
#pragma unroll
  for (int mt = 0; mt < 2; ++mt) {
    const __bf16* qrow = qbase + (size_t)(mt * 16 + m) * DHEAD;
    aQ[mt][0] = load_a_frag(qrow, 0, half);
    aQ[mt][1] = load_a_frag(qrow, 32, half);
  }

  v8f acc[2][4];
#pragma unroll
  for (int mt = 0; mt < 2; ++mt)
#pragma unroll
    for (int db = 0; db < 4; ++db) acc[mt][db] = zero8();
  float m_i = -1.0e30f;  // running max   (row = lane)
  float l_i = 0.0f;      // running denom (row = lane)

  issue_tile(0, 0);  // prologue: prefetch tile 0 into buffer 0

  for (int kt = 0; kt < NTILES; ++kt) {
    const int kb = kt * KTILE;
    const int buf = kt & 1;

    // Overlap: launch next tile's DMA, then wait for this tile's 16 transfers
    // (in-order completion -> outstanding <= 16 means tile kt has landed).
    if (kt + 1 < NTILES) {
      issue_tile(kt + 1, buf ^ 1);
      asm volatile("s_wait_asynccnt 0x10" ::: "memory");
    } else {
      asm volatile("s_wait_asynccnt 0x0" ::: "memory");
    }
    __syncthreads();

    // ---- S = Q . K^T : 32 q-rows x 32 keys (K fragments shared by M-tiles) ----
#pragma unroll
    for (int nb = 0; nb < 2; ++nb) {
      const __bf16* kcol = &sK[buf][(size_t)(nb * 16 + m) * DHEAD];
      v16bf b0 = load_b_frag(kcol + 16 * half);        // contraction d = 0..31
      v16bf b1 = load_b_frag(kcol + 32 + 16 * half);   // contraction d = 32..63
#pragma unroll
      for (int mt = 0; mt < 2; ++mt) {
        v8f c = zero8();
        c = wmma_bf16(aQ[mt][0], b0, c);
        c = wmma_bf16(aQ[mt][1], b1, c);
#pragma unroll
        for (int r = 0; r < 8; ++r)
          sS[(mt * 16 + r + 8 * half) * 32 + nb * 16 + m] = c[r];
      }
    }
    __syncthreads();

    // ---- online softmax, one lane per q-row (32 rows, no divergence) ----
    {
      float sv[32];
      float tmax = -1.0e30f;
#pragma unroll
      for (int j = 0; j < 32; ++j) {
        float s = sS[lane * 32 + j] * scale;
        if (kb + j >= vlen) s = MASK_VAL;  // padding mask on scaled scores
        sv[j] = s;
        tmax = fmaxf(tmax, s);
      }
      const float mnew = fmaxf(m_i, tmax);
      const float corr = __expf(m_i - mnew);
      float sum = 0.0f;
#pragma unroll
      for (int j = 0; j < 32; ++j) {
        float p = __expf(sv[j] - mnew);
        sP[lane * 32 + j] = (__bf16)p;
        sum += p;
      }
      l_i = l_i * corr + sum;
      m_i = mnew;
      sRow[lane] = corr;
    }
    __syncthreads();

    // ---- rescale accumulators ----
#pragma unroll
    for (int mt = 0; mt < 2; ++mt) {
      float corr[8];
#pragma unroll
      for (int r = 0; r < 8; ++r) corr[r] = sRow[mt * 16 + r + 8 * half];
#pragma unroll
      for (int db = 0; db < 4; ++db)
#pragma unroll
        for (int r = 0; r < 8; ++r) acc[mt][db][r] *= corr[r];
    }

    // ---- acc += P . V (V fragments from LDS, shared by both M-tiles) ----
    v16bf bv[4];
#pragma unroll
    for (int db = 0; db < 4; ++db) {
      const __bf16* vp = &sV[buf][(size_t)(16 * half) * DHEAD + db * 16 + m];
#pragma unroll
      for (int e = 0; e < 16; ++e) bv[db][e] = vp[(size_t)e * DHEAD];
    }
#pragma unroll
    for (int mt = 0; mt < 2; ++mt) {
      v16bf aP = load_a_frag(&sP[(mt * 16 + m) * 32], 0, half);
#pragma unroll
      for (int db = 0; db < 4; ++db)
        acc[mt][db] = wmma_bf16(aP, bv[db], acc[mt][db]);
    }
    __syncthreads();  // protect sS/sP reuse next iteration
  }

  // ---- normalize and emit [B, Q, H*DH] bf16 ----
  sRow[lane] = 1.0f / l_i;
  __syncthreads();
#pragma unroll
  for (int mt = 0; mt < 2; ++mt) {
    float inv[8];
#pragma unroll
    for (int r = 0; r < 8; ++r) inv[r] = sRow[mt * 16 + r + 8 * half];
#pragma unroll
    for (int db = 0; db < 4; ++db)
#pragma unroll
      for (int r = 0; r < 8; ++r) {
        const int qrow = qt * 32 + mt * 16 + r + 8 * half;
        const int d = db * 16 + m;
        attn_out[((size_t)b * SEQ + qrow) * DMODEL + h * DHEAD + d] =
            (__bf16)(acc[mt][db][r] * inv[r]);
      }
  }
}

extern "C" void kernel_launch(void* const* d_in, const int* in_sizes, int n_in,
                              void* d_out, int out_size, void* d_ws, size_t ws_size,
                              hipStream_t stream) {
  const float* q  = (const float*)d_in[0];
  const float* k  = (const float*)d_in[1];
  const float* v  = (const float*)d_in[2];
  const int* vlen = (const int*)d_in[3];
  const float* W_q = (const float*)d_in[4];
  const float* W_k = (const float*)d_in[5];
  const float* W_v = (const float*)d_in[6];
  const float* W_o = (const float*)d_in[7];
  float* out = (float*)d_out;

  constexpr size_t XIN = (size_t)BATCH * SEQ * DMODEL;  // 4,194,304 elements
  constexpr size_t WSZ = (size_t)DMODEL * DMODEL;       //   262,144 elements

  __bf16* ws = (__bf16*)d_ws;
  __bf16* qb = ws;
  __bf16* kb = qb + XIN;
  __bf16* vb = kb + XIN;
  __bf16* wq = vb + XIN;
  __bf16* wk = wq + WSZ;
  __bf16* wv = wk + WSZ;
  __bf16* wo = wv + WSZ;
  __bf16* Qp = wo + WSZ;      // projected, split-head [B,H,S,DH]
  __bf16* Kp = Qp + XIN;
  __bf16* Vp = Kp + XIN;
  __bf16* attnb = Vp + XIN;   // attention output [B,Q,D] bf16
  // total ~58 MB of workspace

  const int nx = (int)XIN, nw = (int)WSZ;
  cvt_f32_bf16_kernel<<<(nx + 255) / 256, 256, 0, stream>>>(q, qb, nx);
  cvt_f32_bf16_kernel<<<(nx + 255) / 256, 256, 0, stream>>>(k, kb, nx);
  cvt_f32_bf16_kernel<<<(nx + 255) / 256, 256, 0, stream>>>(v, vb, nx);
  cvt_f32_bf16_kernel<<<(nw + 255) / 256, 256, 0, stream>>>(W_q, wq, nw);
  cvt_f32_bf16_kernel<<<(nw + 255) / 256, 256, 0, stream>>>(W_k, wk, nw);
  cvt_f32_bf16_kernel<<<(nw + 255) / 256, 256, 0, stream>>>(W_v, wv, nw);
  cvt_f32_bf16_kernel<<<(nw + 255) / 256, 256, 0, stream>>>(W_o, wo, nw);

  dim3 ggemm(DMODEL / 64, (BATCH * SEQ) / 16);  // 8 x 512 (16x64 tiles)
  gemm_xwT_kernel<<<ggemm, 32, 0, stream>>>(qb, wq, Qp, nullptr, 0);
  gemm_xwT_kernel<<<ggemm, 32, 0, stream>>>(kb, wk, Kp, nullptr, 0);
  gemm_xwT_kernel<<<ggemm, 32, 0, stream>>>(vb, wv, Vp, nullptr, 0);

  dim3 gattn(SEQ / 32, BATCH * NHEAD);          // 64 x 32 waves
  flash_attn_kernel<<<gattn, 32, 0, stream>>>(Qp, Kp, Vp, vlen, attnb);

  gemm_xwT_kernel<<<ggemm, 32, 0, stream>>>(attnb, wo, nullptr, out, 1);
}
